// Mask_demo_13477607375176
// MI455X (gfx1250) — matrix operand order repaired and checked by reading, not compiled, not run on hardware
//
#include <hip/hip_runtime.h>

// ---------------------------------------------------------------------------
// Bit-plane SR net on gfx1250 via v_wmma_f32_16x16x32_f16.
// Rotation ensemble folded into rotated 5x5 kernels (pointwise tail commutes
// with spatial rotation), 5x5 conv done as im2col GEMM (K=25 padded to 32),
// 1x1 conv chain done as 64x64 GEMMs per 16-pixel M-tile.
// ---------------------------------------------------------------------------

typedef _Float16 half_t;
typedef __attribute__((ext_vector_type(16))) _Float16 v16h;
typedef __attribute__((ext_vector_type(8)))  float    v8f;
typedef unsigned int u32;
typedef __attribute__((ext_vector_type(4))) u32 u32x4;

union AFrag { v16h h; u32 u[8]; };
union BFrag { v16h h; u32x4 q[2]; };

#define ACT_STRIDE 68            // 64 channels + pad (bank-conflict avoidance)
#define IM_STRIDE  36            // 32 taps + pad
#define PLANE_HALVES (50 * 512)  // 16 conv frags + 32 layer frags + 2 W6 frags
#define BIAS_OFF_BYTES 102400    // 2 * PLANE_HALVES * 2 bytes
#define BIAS_PER_PLANE 336       // b1(64) + b2..b5(256) + b6(16)

__device__ __forceinline__ v8f wmma32(v16h a, v16h b, v8f c) {
  return __builtin_amdgcn_wmma_f32_16x16x32_f16(false, a, false, b,
                                                (short)0, c, false, false);
}

// Weights are pre-swizzled so each lane's 16 halves are contiguous:
// element (lane*16 + i) holds B[k = 16*(lane>>4) + i][n = lane&15].
__device__ __forceinline__ BFrag load_bfrag(const half_t* __restrict__ frag,
                                            int lane) {
  BFrag b;
  const u32x4* q = (const u32x4*)(frag + lane * 16);
  b.q[0] = q[0];
  b.q[1] = q[1];
  return b;
}

// ---------------------------------------------------------------------------
// Prep: build per-plane weight blob in WMMA B-fragment layout + bias table.
// frag ids (per plane): [0..15]  conv: r*4 + ntile     (K=32: 25 taps + 0 pad)
//                       [16..47] layers W2..W5: 16 + (L*4+nt)*2 + kt
//                       [48..49] W6: 48 + kt
// rotated kernel W_r = rot^{-r}(W*mask):
//   r0:(a,b)  r1:(4-b,a)  r2:(4-a,4-b)  r3:(b,4-a)
// ---------------------------------------------------------------------------
__global__ void prep_weights(const float* __restrict__ W1, const float* __restrict__ b1,
                             const float* __restrict__ W2, const float* __restrict__ b2,
                             const float* __restrict__ W3, const float* __restrict__ b3,
                             const float* __restrict__ W4, const float* __restrict__ b4,
                             const float* __restrict__ W5, const float* __restrict__ b5,
                             const float* __restrict__ W6, const float* __restrict__ b6,
                             const float* __restrict__ mask,
                             half_t* __restrict__ blob, float* __restrict__ biases,
                             int plane) {
  const int tid = threadIdx.x;
  half_t* bp = blob + plane * PLANE_HALVES;
  const float* Wm[4] = {W2, W3, W4, W5};

  for (int e = tid; e < PLANE_HALVES; e += 256) {
    int f = e >> 9, s = e & 511;
    int lane = s >> 4, i = s & 15;
    int nn = lane & 15;
    int k = ((lane >> 4) << 4) | i;  // 0..31
    float val = 0.0f;
    if (f < 16) {
      int r = f >> 2, nt = f & 3, ch = nt * 16 + nn;
      if (k < 25) {
        int a = k / 5, b = k % 5, sa, sb;
        switch (r) {
          case 0:  sa = a;     sb = b;     break;
          case 1:  sa = 4 - b; sb = a;     break;
          case 2:  sa = 4 - a; sb = 4 - b; break;
          default: sa = b;     sb = 4 - a; break;
        }
        val = W1[ch * 25 + sa * 5 + sb] * mask[sa * 5 + sb];
      }
    } else if (f < 48) {
      int g = f - 16;
      int kt = g & 1, nt = (g >> 1) & 3, L = g >> 3;
      val = Wm[L][(nt * 16 + nn) * 64 + kt * 32 + k];
    } else {
      int kt = f - 48;
      val = W6[nn * 64 + kt * 32 + k];
    }
    bp[e] = (half_t)val;
  }

  float* bb = biases + plane * BIAS_PER_PLANE;
  const float* bl[4] = {b2, b3, b4, b5};
  for (int t2 = tid; t2 < BIAS_PER_PLANE; t2 += 256) {
    float v;
    if (t2 < 64)       v = b1[t2];
    else if (t2 < 320) v = bl[(t2 - 64) >> 6][(t2 - 64) & 63];
    else               v = b6[t2 - 320];
    bb[t2] = v;
  }
}

// ---------------------------------------------------------------------------
// Main: one workgroup per (image, 16x16 LR tile). 8 waves x 32 pixels.
// ---------------------------------------------------------------------------
__global__ __launch_bounds__(256) void sr_main(const float* __restrict__ img,
                                               const half_t* __restrict__ blob,
                                               const float* __restrict__ biases,
                                               float* __restrict__ out) {
  __shared__ half_t s_act[256 * ACT_STRIDE];   // activations [pixel][64ch]
  __shared__ half_t s_im[256 * IM_STRIDE];     // im2col      [pixel][32taps]
  __shared__ float  s_tile[3][400];            // 20x20 halo: x, msb, lsb

  const int tid  = threadIdx.x;
  const int lane = tid & 31;
  const int wv   = tid >> 5;
  const int hl   = lane >> 4;   // lane half (A: K-group, B/D: row group)
  const int ln   = lane & 15;   // A: M row, B/D: N column
  const int mrow0 = wv * 32;

  const int blk = blockIdx.x;
  const int n   = blk / 36;
  const int til = blk % 36;
  const int ty = til / 6, tx = til % 6;
  const float* imgN = img + n * 96 * 96;

  // --- bit-plane slicing of halo'd input tile (zero pad outside image) ---
  for (int e = tid; e < 400; e += 256) {
    int yy = e / 20, xx = e % 20;
    int gy = ty * 16 - 2 + yy;
    int gx = tx * 16 - 2 + xx;
    float v = 0.0f;
    if (gy >= 0 && gy < 96 && gx >= 0 && gx < 96) v = imgN[gy * 96 + gx];
    float L   = floorf(v * 255.0f);
    float m16 = floorf(L * (1.0f / 16.0f)) * 16.0f;
    float lsb = (L - m16) * 16.0f;
    s_tile[0][e] = v;
    s_tile[1][e] = m16 * (1.0f / 255.0f);
    s_tile[2][e] = lsb * (1.0f / 255.0f);
  }
  __syncthreads();

  v8f accM[2], accL[2];

  for (int plane = 0; plane < 2; ++plane) {
    const half_t* wb = blob + plane * PLANE_HALVES;
    const float*  bs = biases + plane * BIAS_PER_PLANE;

    // im2col build: thread t owns pixel t (rows are wave-private downstream)
    {
      int p = tid;
      int py = p >> 4, px = p & 15;
      half_t* row = s_im + p * IM_STRIDE;
#pragma unroll
      for (int a = 0; a < 5; ++a)
#pragma unroll
        for (int b = 0; b < 5; ++b)
          row[a * 5 + b] = (half_t)s_tile[plane + 1][(py + a) * 20 + (px + b)];
#pragma unroll
      for (int k = 25; k < 32; ++k) row[k] = (half_t)0.0f;
    }

    v8f acc[2];
    acc[0] = (v8f){};
    acc[1] = (v8f){};

    for (int r = 0; r < 4; ++r) {
      // ---- 5x5 conv as single-K-tile WMMA from im2col ----
      AFrag ca[2];
#pragma unroll
      for (int mt = 0; mt < 2; ++mt) {
        const half_t* base = s_im + (mrow0 + mt * 16 + ln) * IM_STRIDE;
#pragma unroll
        for (int j = 0; j < 8; ++j) {
          int k0 = 16 * hl + 2 * j;
          ca[mt].u[j] = *(const u32*)(base + k0);
        }
      }
#pragma unroll
      for (int nt = 0; nt < 4; ++nt) {
        BFrag bf = load_bfrag(wb + (r * 4 + nt) * 512, lane);
        float bv = bs[nt * 16 + ln];
#pragma unroll
        for (int mt = 0; mt < 2; ++mt) {
          v8f c = {bv, bv, bv, bv, bv, bv, bv, bv};
          c = wmma32(ca[mt].h, bf.h, c);
#pragma unroll
          for (int j = 0; j < 8; ++j)
            s_act[(mrow0 + mt * 16 + j + 8 * hl) * ACT_STRIDE + nt * 16 + ln] =
                (half_t)fmaxf(c[j], 0.0f);  // ReLU feeds W2
        }
      }

      // ---- four 64x64 pointwise layers (W2..W5), in-place in s_act ----
      for (int L = 0; L < 4; ++L) {
        AFrag af[2][2];
#pragma unroll
        for (int mt = 0; mt < 2; ++mt) {
          const half_t* rbase = s_act + (mrow0 + mt * 16 + ln) * ACT_STRIDE;
#pragma unroll
          for (int kt = 0; kt < 2; ++kt)
#pragma unroll
            for (int j = 0; j < 8; ++j) {
              int k0 = kt * 32 + 16 * hl + 2 * j;
              af[mt][kt].u[j] = *(const u32*)(rbase + k0);
            }
        }
        const float*  bLp = bs + 64 + L * 64;
        const half_t* wLp = wb + (16 + L * 8) * 512;
        const bool relu = (L < 3);  // W5 output feeds W6 without ReLU
#pragma unroll
        for (int nt = 0; nt < 4; ++nt) {
          BFrag b0 = load_bfrag(wLp + (nt * 2 + 0) * 512, lane);
          BFrag b1 = load_bfrag(wLp + (nt * 2 + 1) * 512, lane);
          float bv = bLp[nt * 16 + ln];
#pragma unroll
          for (int mt = 0; mt < 2; ++mt) {
            v8f c = {bv, bv, bv, bv, bv, bv, bv, bv};
            c = wmma32(af[mt][0].h, b0.h, c);
            c = wmma32(af[mt][1].h, b1.h, c);
#pragma unroll
            for (int j = 0; j < 8; ++j) {
              float v = c[j];
              if (relu) v = fmaxf(v, 0.0f);
              s_act[(mrow0 + mt * 16 + j + 8 * hl) * ACT_STRIDE + nt * 16 + ln] =
                  (half_t)v;
            }
          }
        }
      }

      // ---- W6: 64 -> 16, then floor(127*) ensemble accumulate ----
      {
        AFrag af[2][2];
#pragma unroll
        for (int mt = 0; mt < 2; ++mt) {
          const half_t* rbase = s_act + (mrow0 + mt * 16 + ln) * ACT_STRIDE;
#pragma unroll
          for (int kt = 0; kt < 2; ++kt)
#pragma unroll
            for (int j = 0; j < 8; ++j) {
              int k0 = kt * 32 + 16 * hl + 2 * j;
              af[mt][kt].u[j] = *(const u32*)(rbase + k0);
            }
        }
        BFrag b0 = load_bfrag(wb + 48 * 512, lane);
        BFrag b1 = load_bfrag(wb + 49 * 512, lane);
        float bv = bs[320 + ln];
#pragma unroll
        for (int mt = 0; mt < 2; ++mt) {
          v8f c = {bv, bv, bv, bv, bv, bv, bv, bv};
          c = wmma32(af[mt][0].h, b0.h, c);
          c = wmma32(af[mt][1].h, b1.h, c);
#pragma unroll
          for (int j = 0; j < 8; ++j)
            acc[mt][j] += floorf(127.0f * c[j]);
        }
      }
    }  // rotations

    if (plane == 0) { accM[0] = acc[0]; accM[1] = acc[1]; }
    else            { accL[0] = acc[0]; accL[1] = acc[1]; }
  }  // planes

  // --- combine planes, pixel-shuffle x4, nearest-upsample add, clip ---
#pragma unroll
  for (int mt = 0; mt < 2; ++mt) {
#pragma unroll
    for (int j = 0; j < 8; ++j) {
      float mo = fminf(fmaxf(accM[mt][j] * (1.0f / 765.0f), -1.0f), 1.0f);
      float lo = fminf(fmaxf(accL[mt][j] * (1.0f / 510.0f), -1.0f), 1.0f);
      int p  = mrow0 + mt * 16 + j + 8 * hl;  // D-layout pixel row
      int py = p >> 4, px = p & 15;
      float base = s_tile[0][(py + 2) * 20 + (px + 2)];
      float v = fminf(fmaxf(mo + lo + base, 0.0f), 1.0f);
      int hy = (ty * 16 + py) * 4 + (ln >> 2);  // channel c = ln = r1*4+r2
      int hx = (tx * 16 + px) * 4 + (ln & 3);
      out[(size_t)n * (384 * 384) + (size_t)hy * 384 + hx] = v;
    }
  }
}

// ---------------------------------------------------------------------------
// d_in layout (tree-flattened setup_inputs): 0=img_lr, 1=msb_mask, 2=lsb_mask,
// 3..14 = msb_params (W1,b1,W2,b2,W3,b3,W4,b4,W5,b5,W6,b6), 15..26 = lsb_params
// Workspace: [0,102400) weight blob (f16, fragment-swizzled), then 672 f32 bias
// ---------------------------------------------------------------------------
extern "C" void kernel_launch(void* const* d_in, const int* in_sizes, int n_in,
                              void* d_out, int out_size, void* d_ws, size_t ws_size,
                              hipStream_t stream) {
  (void)in_sizes; (void)n_in; (void)out_size; (void)ws_size;
  const float* img   = (const float*)d_in[0];
  const float* mmask = (const float*)d_in[1];
  const float* lmask = (const float*)d_in[2];
  const float* m[12];
  const float* l[12];
  for (int i = 0; i < 12; ++i) {
    m[i] = (const float*)d_in[3 + i];
    l[i] = (const float*)d_in[15 + i];
  }
  half_t* blob  = (half_t*)d_ws;
  float* biases = (float*)((char*)d_ws + BIAS_OFF_BYTES);

  prep_weights<<<1, 256, 0, stream>>>(m[0], m[1], m[2], m[3], m[4], m[5],
                                      m[6], m[7], m[8], m[9], m[10], m[11],
                                      mmask, blob, biases, 0);
  prep_weights<<<1, 256, 0, stream>>>(l[0], l[1], l[2], l[3], l[4], l[5],
                                      l[6], l[7], l[8], l[9], l[10], l[11],
                                      lmask, blob, biases, 1);

  // 12 images x 36 tiles of 16x16 LR pixels
  sr_main<<<432, 256, 0, stream>>>(img, blob, biases, (float*)d_out);
}
